// PoincareAttentionLayer_56203942035758
// MI455X (gfx1250) — matrix operand order repaired and checked by reading, not compiled, not run on hardware
//
#include <hip/hip_runtime.h>
#include <math.h>

// ---------------------------------------------------------------------------
// Types for CDNA5 WMMA (wave32)
// ---------------------------------------------------------------------------
typedef __attribute__((ext_vector_type(16))) __bf16 v16bf;
typedef __attribute__((ext_vector_type(8)))  __bf16 v8bf;
typedef __attribute__((ext_vector_type(8)))  float  v8f;

#define EPSF  1e-15f
#define CLIPF (1.0f - 1e-5f)
#define SQRT_HALF 0.70710678118654752f

union BF16Frag { v16bf v; v8bf h[2]; };

// ---------------------------------------------------------------------------
// CDNA5 async global->LDS DMA (ASYNCcnt-tracked; no VGPR round trip).
// LDS destination address = low 32 bits of the generic shared-aperture addr.
// ---------------------------------------------------------------------------
__device__ inline void async_ld_b128(unsigned lds_addr, const void* g) {
  asm volatile("global_load_async_to_lds_b128 %0, %1, off"
               :: "v"(lds_addr), "v"((unsigned long long)g)
               : "memory");
}
__device__ inline void wait_async0() {
  asm volatile("s_wait_asynccnt 0x0" ::: "memory");
}

// ---------------------------------------------------------------------------
// Block reductions (256 threads)
// ---------------------------------------------------------------------------
__device__ inline float block_reduce_sum(float v, float* sm) {
  const int t = threadIdx.x;
  sm[t] = v;
  __syncthreads();
  for (int s = 128; s > 0; s >>= 1) {
    if (t < s) sm[t] += sm[t + s];
    __syncthreads();
  }
  float r = sm[0];
  __syncthreads();
  return r;
}

__device__ inline float block_reduce_max(float v, float* sm) {
  const int t = threadIdx.x;
  sm[t] = v;
  __syncthreads();
  for (int s = 128; s > 0; s >>= 1) {
    if (t < s) sm[t] = fmaxf(sm[t], sm[t + s]);
    __syncthreads();
  }
  float r = sm[0];
  __syncthreads();
  return r;
}

// ---------------------------------------------------------------------------
// Generic batched bf16 GEMM:  C[M,N] = A[M,K] (row-major) * Bt[N,K]^T
// Block = 256 thr = 8 waves; block tile = 128 (M) x 64 (N); K step = 32.
// B tile DMA'd into double-buffered LDS via global_load_async_to_lds_b128;
// fragments per CDNA5 WMMA layout.
// ---------------------------------------------------------------------------
__global__ __launch_bounds__(256) void gemm_bf16_tn(
    const __bf16* __restrict__ A,   // [M,K] row-major (per batch)
    const __bf16* __restrict__ Bt,  // [N,K] row-major (per batch)
    float* __restrict__ C,          // [M,N] (per batch)
    int M, int N, int K,
    long long strideA, long long strideB, long long strideC)
{
  (void)M;
  __shared__ __bf16 lds[2][64 * 32];

  const int bz = blockIdx.z;
  A  += (long long)bz * strideA;
  Bt += (long long)bz * strideB;
  C  += (long long)bz * strideC;

  const int n0   = blockIdx.x * 64;
  const int m0   = blockIdx.y * 128;
  const int tid  = threadIdx.x;
  const int wave = tid >> 5;
  const int lane = tid & 31;
  const int half = lane >> 4;    // 0: lanes 0-15, 1: lanes 16-31
  const int l15  = lane & 15;
  const int mrow = m0 + wave * 16 + l15;   // A row owned by this lane

  // cooperative Bt tile mapping: 64 rows x 32 halfs, 16B per thread
  const int tn = tid >> 2;        // 0..63
  const int tk = (tid & 3) * 8;   // 0,8,16,24

  const unsigned lds0 = (unsigned)(unsigned long long)&lds[0][tn * 32 + tk];
  const unsigned lds1 = (unsigned)(unsigned long long)&lds[1][tn * 32 + tk];

  v8f acc0 = {}; v8f acc1 = {}; v8f acc2 = {}; v8f acc3 = {};

  const int nIter = K >> 5;

  // prologue: DMA first Bt tile into LDS buffer 0
  async_ld_b128(lds0, Bt + (long long)(n0 + tn) * K + tk);
  wait_async0();
  __syncthreads();

  for (int it = 0; it < nIter; ++it) {
    const int cur = it & 1;
    const int k0  = it << 5;

    if (it + 1 < nIter) {
      // DMA next Bt tile into the other LDS buffer; overlaps the WMMAs below.
      async_ld_b128(cur ? lds0 : lds1,
                    Bt + (long long)(n0 + tn) * K + (k0 + 32) + tk);
      if (k0 + 64 < K)
        __builtin_prefetch(A + (long long)mrow * K + k0 + 64, 0, 1); // global_prefetch_b8
    }

    // A fragment (16x32 bf16): lanes 0-15 hold K {k0..k0+7, k0+16..k0+23},
    // lanes 16-31 hold K {k0+8..k0+15, k0+24..k0+31}
    BF16Frag fa;
    const __bf16* pa = A + (long long)mrow * K + k0 + half * 8;
    fa.h[0] = *(const v8bf*)pa;
    fa.h[1] = *(const v8bf*)(pa + 16);

    // 4 B fragments (32x16 each) from LDS: lane holds column n, K {0..7, 16..23} (lo half)
    // or K {8..15, 24..31} (hi half), matching the A fragment K order.
    const __bf16* lb = &lds[cur][0];
    BF16Frag fb[4];
#pragma unroll
    for (int j = 0; j < 4; ++j) {
      const __bf16* p = lb + (j * 16 + l15) * 32 + half * 8;
      fb[j].h[0] = *(const v8bf*)p;        // ds_load_b128
      fb[j].h[1] = *(const v8bf*)(p + 16); // ds_load_b128
    }

    acc0 = __builtin_amdgcn_wmma_f32_16x16x32_bf16(false, fa.v, false, fb[0].v, (short)0, acc0, false, false);
    acc1 = __builtin_amdgcn_wmma_f32_16x16x32_bf16(false, fa.v, false, fb[1].v, (short)0, acc1, false, false);
    acc2 = __builtin_amdgcn_wmma_f32_16x16x32_bf16(false, fa.v, false, fb[2].v, (short)0, acc2, false, false);
    acc3 = __builtin_amdgcn_wmma_f32_16x16x32_bf16(false, fa.v, false, fb[3].v, (short)0, acc3, false, false);

    // All async DMA for the next buffer must land before any wave reads it.
    wait_async0();
    __syncthreads();
  }

  // Store: C/D layout -> VGPR r: lanes 0-15 row m0+w*16+r, lanes 16-31 row +8
  const int rbase = m0 + wave * 16 + half * 8;
#pragma unroll
  for (int r = 0; r < 8; ++r) {
    float* crow = C + (long long)(rbase + r) * N + n0 + l15;
    crow[0]  = acc0[r];
    crow[16] = acc1[r];
    crow[32] = acc2[r];
    crow[48] = acc3[r];
  }
}

// ---------------------------------------------------------------------------
// Weight prep: zn[n] = ||W[:,n]||, Wt[n,k] = bf16(W[k,n]/zn[n])
// ---------------------------------------------------------------------------
__global__ __launch_bounds__(256) void prep_weight(
    const float* __restrict__ W, __bf16* __restrict__ Wt,
    float* __restrict__ zn, int Kd, int Nd)
{
  __shared__ float red[256];
  const int n = blockIdx.x;
  float ss = 0.f;
  for (int k = threadIdx.x; k < Kd; k += 256) {
    float v = W[(long long)k * Nd + n];
    ss += v * v;
  }
  ss = block_reduce_sum(ss, red);
  const float znv = sqrtf(fmaxf(ss, EPSF));
  if (threadIdx.x == 0) zn[n] = znv;
  const float inv = 1.0f / znv;
  for (int k = threadIdx.x; k < Kd; k += 256)
    Wt[(long long)n * Kd + k] = (__bf16)(W[(long long)k * Nd + n] * inv);
}

// ---------------------------------------------------------------------------
// f32 rows -> bf16 copy + row sum of squares (D multiple of 256)
// ---------------------------------------------------------------------------
__global__ __launch_bounds__(256) void rows_to_bf16_sumsq(
    const float* __restrict__ src, __bf16* __restrict__ dst,
    float* __restrict__ sumsq, int D)
{
  __shared__ float red[256];
  const long long row = blockIdx.x;
  float ss = 0.f;
  for (int d = threadIdx.x; d < D; d += 256) {
    float v = src[row * D + d];
    ss += v * v;
    dst[row * D + d] = (__bf16)v;
  }
  ss = block_reduce_sum(ss, red);
  if (threadIdx.x == 0) sumsq[row] = ss;
}

__global__ __launch_bounds__(256) void row_sumsq_bf16(
    const __bf16* __restrict__ src, float* __restrict__ out, int D)
{
  __shared__ float red[256];
  const long long row = blockIdx.x;
  float ss = 0.f;
  for (int d = threadIdx.x; d < D; d += 256) {
    float v = (float)src[row * D + d];
    ss += v * v;
  }
  ss = block_reduce_sum(ss, red);
  if (threadIdx.x == 0) out[row] = ss;
}

// ---------------------------------------------------------------------------
// LDS-tiled transpose f32 -> bf16 with optional per-source-row scale
// src [R, Cc] -> dst [Cc, R] per batch; rscale indexed [b*R + r] or nullptr
// ---------------------------------------------------------------------------
__global__ __launch_bounds__(256) void transpose_scale_f32_bf16(
    const float* __restrict__ src, __bf16* __restrict__ dst,
    const float* __restrict__ rscale, int R, int Cc)
{
  __shared__ float tile[32][33];
  const int b = blockIdx.z;
  const float* s = src + (long long)b * R * Cc;
  __bf16*      d = dst + (long long)b * R * Cc;
  const int c0 = blockIdx.x * 32;
  const int r0 = blockIdx.y * 32;
  const int tx = threadIdx.x & 31;
  const int ty = threadIdx.x >> 5;  // 0..7
#pragma unroll
  for (int i = 0; i < 4; ++i) {
    const int r = r0 + ty + i * 8;
    const float sc = rscale ? rscale[(long long)b * R + r] : 1.0f;
    tile[ty + i * 8][tx] = s[(long long)r * Cc + c0 + tx] * sc;
  }
  __syncthreads();
#pragma unroll
  for (int i = 0; i < 4; ++i) {
    const int c = c0 + ty + i * 8;
    d[(long long)c * R + r0 + tx] = (__bf16)tile[tx][ty + i * 8];
  }
}

// ---------------------------------------------------------------------------
// gamma[b,s] = 2 / max(1 - ||enc_b[b,s]||^2, EPS);  gm1 = gamma - 1
// ---------------------------------------------------------------------------
__global__ __launch_bounds__(256) void gamma_kernel(
    const float* __restrict__ encb, float* __restrict__ gamma,
    float* __restrict__ gm1, int E)
{
  __shared__ float red[256];
  const long long row = blockIdx.x;
  float ss = 0.f;
  for (int e = threadIdx.x; e < E; e += 256) {
    float v = encb[row * E + e];
    ss += v * v;
  }
  ss = block_reduce_sum(ss, red);
  const float g = 2.0f / fmaxf(1.0f - ss, EPSF);
  if (threadIdx.x == 0) { gamma[row] = g; gm1[row] = g - 1.0f; }
}

// ---------------------------------------------------------------------------
// sb[b] = S - sum(mask[b,:])
// ---------------------------------------------------------------------------
__global__ __launch_bounds__(256) void mask_count(
    const unsigned char* __restrict__ mask, float* __restrict__ sb, int S)
{
  __shared__ float red[256];
  const int b = blockIdx.x;
  float c = 0.f;
  for (int s = threadIdx.x; s < S; s += 256)
    c += mask[(long long)b * S + s] ? 1.0f : 0.0f;
  c = block_reduce_sum(c, red);
  if (threadIdx.x == 0) sb[b] = (float)S - c;
}

// ---------------------------------------------------------------------------
// q = mobius_scalar_mul(sqrt(.5), mobius_add(tgt, poincare_linear(x,...)))
// One block per (b,t) row; E = 768 = 3 * 256.
// ---------------------------------------------------------------------------
__global__ __launch_bounds__(256) void poincare_q_kernel(
    const float* __restrict__ xw,   // [row, E] = x @ zu
    const float* __restrict__ xn2,  // [row]   = ||x||^2
    const float* __restrict__ zn,   // [E]
    const float* __restrict__ bias, // [E]
    const float* __restrict__ tgt,  // [row, E]
    __bf16* __restrict__ qbf, float* __restrict__ qn2, int E)
{
  __shared__ float red[256];
  const long long row = blockIdx.x;
  const float cx2  = xn2[row];                       // C = 1
  const float invd = 1.0f / fmaxf(1.0f - cx2, EPSF);

  float wv[3], tv[3];
  float sw2 = 0.f, stw = 0.f, st2 = 0.f;
#pragma unroll
  for (int i = 0; i < 3; ++i) {
    const int e = threadIdx.x + i * 256;
    const float r   = bias[e];
    const float xwv = xw[row * E + e];
    const float num = 2.0f * xwv * coshf(2.0f * r) - (1.0f + cx2) * sinhf(2.0f * r);
    const float y   = 2.0f * zn[e] * asinhf(num * invd);
    const float w   = sinhf(y);
    const float t   = tgt[row * E + e];
    wv[i] = w; tv[i] = t;
    sw2 += w * w; stw += t * w; st2 += t * t;
  }
  sw2 = block_reduce_sum(sw2, red);
  stw = block_reduce_sum(stw, red);
  st2 = block_reduce_sum(st2, red);

  const float s  = 1.0f / (1.0f + sqrtf(1.0f + sw2));  // pl = s * w
  const float p2 = s * s * sw2;
  const float tp = s * stw;
  const float a1 = 1.0f + 2.0f * tp + p2;
  const float a2 = 1.0f - st2;
  const float im = 1.0f / fmaxf(1.0f + 2.0f * tp + st2 * p2, EPSF);

  float ma[3]; float sm2 = 0.f;
#pragma unroll
  for (int i = 0; i < 3; ++i) {
    ma[i] = (a1 * tv[i] + a2 * s * wv[i]) * im;
    sm2 += ma[i] * ma[i];
  }
  sm2 = block_reduce_sum(sm2, red);

  const float n2 = fmaxf(sm2, EPSF);
  const float n  = sqrtf(n2);
  const float f  = tanhf(SQRT_HALF * atanhf(fminf(n, CLIPF))) / n;
#pragma unroll
  for (int i = 0; i < 3; ++i) {
    const int e = threadIdx.x + i * 256;
    qbf[row * E + e] = (__bf16)(f * ma[i]);
  }
  if (threadIdx.x == 0) qn2[row] = f * f * n2;
}

// ---------------------------------------------------------------------------
// scores = -dist(q, enc_a)/exp(scale); mask; softmax; emit attn (f32 + bf16)
// and midpoint denominator row sum attn . (gamma-1). S = 1024 = 4 * 256.
// ---------------------------------------------------------------------------
__global__ __launch_bounds__(256) void attn_softmax_kernel(
    const float* __restrict__ xy,   // [row, S] = q . enc_a
    const float* __restrict__ qn2,  // [row]
    const float* __restrict__ y2,   // [b, S]
    const unsigned char* __restrict__ mask,  // [b, S]
    const float* __restrict__ gm1,  // [b, S]
    const float* __restrict__ scale,
    float* __restrict__ attn_f32, __bf16* __restrict__ attn_bf,
    float* __restrict__ denrow, int S, int T)
{
  __shared__ float red[256];
  const long long row = blockIdx.x;
  const int b = (int)(row / T);
  const float x2 = qn2[row];
  const float inv_es = expf(-scale[0]);

  float sco[4];
  float mx = -INFINITY;
#pragma unroll
  for (int i = 0; i < 4; ++i) {
    const int sI  = threadIdx.x + i * 256;
    const float xyv = xy[row * S + sI];
    const float yy  = y2[(long long)b * S + sI];
    const float num = fmaxf(x2 - 2.0f * xyv + yy, 0.0f);
    const float den = fmaxf(1.0f - 2.0f * xyv + x2 * yy, EPSF);
    const float m   = sqrtf(num / den);
    float v = -2.0f * atanhf(fminf(m, CLIPF)) * inv_es;
    if (mask[(long long)b * S + sI]) v = -INFINITY;
    sco[i] = v;
    mx = fmaxf(mx, v);
  }
  mx = block_reduce_max(mx, red);

  float se = 0.f;
#pragma unroll
  for (int i = 0; i < 4; ++i) { sco[i] = expf(sco[i] - mx); se += sco[i]; }
  se = block_reduce_sum(se, red);
  const float inv = 1.0f / se;

  float dsum = 0.f;
#pragma unroll
  for (int i = 0; i < 4; ++i) {
    const int sI = threadIdx.x + i * 256;
    const float p = sco[i] * inv;
    attn_f32[row * S + sI] = p;
    attn_bf[row * S + sI]  = (__bf16)p;
    dsum += p * gm1[(long long)b * S + sI];
  }
  dsum = block_reduce_sum(dsum, red);
  if (threadIdx.x == 0) denrow[row] = dsum;
}

// ---------------------------------------------------------------------------
// mid = mobius_scalar_mul(sqrt(sb), mobius_scalar_mul(0.5, nom/den))
// ---------------------------------------------------------------------------
__global__ __launch_bounds__(256) void midpoint_kernel(
    const float* __restrict__ nom, const float* __restrict__ denrow,
    const float* __restrict__ sb, __bf16* __restrict__ midbf,
    float* __restrict__ midn2, int E, int T)
{
  __shared__ float red[256];
  const long long row = blockIdx.x;
  const int b = (int)(row / T);
  float den = denrow[row];
  if (fabsf(den) < 1e-10f) den = 1e-10f;
  const float invden = 1.0f / den;

  float vv[3]; float sv2 = 0.f;
#pragma unroll
  for (int i = 0; i < 3; ++i) {
    const int e = threadIdx.x + i * 256;
    const float v = nom[row * E + e] * invden;
    vv[i] = v; sv2 += v * v;
  }
  sv2 = block_reduce_sum(sv2, red);

  const float n1 = sqrtf(fmaxf(sv2, EPSF));
  const float f1 = tanhf(0.5f * atanhf(fminf(n1, CLIPF))) / n1;
  const float r2 = sqrtf(sb[b]);
  const float nu = fmaxf(f1 * n1, 3.2e-8f);   // ~sqrt(EPS)
  const float f2 = tanhf(r2 * atanhf(fminf(nu, CLIPF))) / nu;
  const float g  = f2 * f1;
#pragma unroll
  for (int i = 0; i < 3; ++i) {
    const int e = threadIdx.x + i * 256;
    midbf[row * E + e] = (__bf16)(g * vv[i]);
  }
  if (threadIdx.x == 0) midn2[row] = g * g * sv2;
}

// ---------------------------------------------------------------------------
// out = mobius_scalar_mul(sqrt(.5), mobius_add(residual, poincare_linear(mid)))
// D = CC = 512 = 2 * 256.
// ---------------------------------------------------------------------------
__global__ __launch_bounds__(256) void final_kernel(
    const float* __restrict__ mw,    // [row, D] = mid @ zu_out
    const float* __restrict__ midn2, // [row]
    const float* __restrict__ zn,    // [D]
    const float* __restrict__ bias,  // [D]
    const float* __restrict__ xin,   // [row, D] residual
    const float* __restrict__ xn2,   // [row] ||residual||^2
    float* __restrict__ out, int D)
{
  __shared__ float red[256];
  const long long row = blockIdx.x;
  const float cx2  = midn2[row];
  const float invd = 1.0f / fmaxf(1.0f - cx2, EPSF);

  float wv[2], xv[2];
  float sw2 = 0.f, sxw = 0.f;
#pragma unroll
  for (int i = 0; i < 2; ++i) {
    const int d = threadIdx.x + i * 256;
    const float r   = bias[d];
    const float mwv = mw[row * D + d];
    const float num = 2.0f * mwv * coshf(2.0f * r) - (1.0f + cx2) * sinhf(2.0f * r);
    const float y   = 2.0f * zn[d] * asinhf(num * invd);
    const float w   = sinhf(y);
    const float x   = xin[row * D + d];
    wv[i] = w; xv[i] = x;
    sw2 += w * w; sxw += x * w;
  }
  sw2 = block_reduce_sum(sw2, red);
  sxw = block_reduce_sum(sxw, red);

  const float t2 = xn2[row];
  const float s  = 1.0f / (1.0f + sqrtf(1.0f + sw2));
  const float p2 = s * s * sw2;
  const float tp = s * sxw;
  const float a1 = 1.0f + 2.0f * tp + p2;
  const float a2 = 1.0f - t2;
  const float im = 1.0f / fmaxf(1.0f + 2.0f * tp + t2 * p2, EPSF);

  float ma[2]; float sm2 = 0.f;
#pragma unroll
  for (int i = 0; i < 2; ++i) {
    ma[i] = (a1 * xv[i] + a2 * s * wv[i]) * im;
    sm2 += ma[i] * ma[i];
  }
  sm2 = block_reduce_sum(sm2, red);

  const float n = sqrtf(fmaxf(sm2, EPSF));
  const float f = tanhf(SQRT_HALF * atanhf(fminf(n, CLIPF))) / n;
#pragma unroll
  for (int i = 0; i < 2; ++i) {
    const int d = threadIdx.x + i * 256;
    out[row * D + d] = f * ma[i];
  }
}

// ---------------------------------------------------------------------------
// Launch
// ---------------------------------------------------------------------------
extern "C" void kernel_launch(void* const* d_in, const int* in_sizes, int n_in,
                              void* d_out, int out_size, void* d_ws, size_t ws_size,
                              hipStream_t stream)
{
  (void)in_sizes; (void)n_in; (void)out_size; (void)ws_size;
  const int B = 16, T = 1024, S = 1024, CC = 512, E = 768;

  const float* x     = (const float*)d_in[0];
  const float* tgt   = (const float*)d_in[1];
  const float* enc_a = (const float*)d_in[2];   // [B,E,S]
  const float* enc_b = (const float*)d_in[3];   // [B,S,E]
  const unsigned char* mask = (const unsigned char*)d_in[4];
  const float* w_in  = (const float*)d_in[5];   // [CC,E]
  const float* b_in  = (const float*)d_in[6];   // [E]
  const float* w_out = (const float*)d_in[7];   // [E,CC]
  const float* b_out = (const float*)d_in[8];   // [CC]
  const float* scale = (const float*)d_in[9];   // [1]

  float* out_main = (float*)d_out;                         // [B,T,CC]
  float* attn_out = (float*)d_out + (size_t)B * T * CC;    // [B,T,S]

  char* ws = (char*)d_ws;
  size_t cur = 0;
  auto alloc = [&](size_t bytes) -> char* {
    char* p = ws + cur;
    cur = (cur + bytes + 255) & ~(size_t)255;
    return p;
  };

  float*  zn_in   = (float*) alloc((size_t)E * 4);
  __bf16* w_inT   = (__bf16*)alloc((size_t)E * CC * 2);
  float*  zn_out  = (float*) alloc((size_t)CC * 4);
  __bf16* w_outT  = (__bf16*)alloc((size_t)CC * E * 2);
  __bf16* x_bf    = (__bf16*)alloc((size_t)B * T * CC * 2);
  float*  xn2     = (float*) alloc((size_t)B * T * 4);
  float*  xw      = (float*) alloc((size_t)B * T * E * 4);   // later reused as nom
  __bf16* q_bf    = (__bf16*)alloc((size_t)B * T * E * 2);   // later reused as mid_bf
  float*  qn2     = (float*) alloc((size_t)B * T * 4);
  __bf16* aT      = (__bf16*)alloc((size_t)B * S * E * 2);   // enc_a transposed [B,S,E]
  float*  y2      = (float*) alloc((size_t)B * S * 4);
  float*  xy      = (float*) alloc((size_t)B * T * S * 4);
  __bf16* attn_bf = (__bf16*)alloc((size_t)B * T * S * 2);   // later reused as mw f32
  float*  mwf     = (float*)attn_bf;                          // [B,T,CC] f32, same bytes
  __bf16* gbT     = (__bf16*)alloc((size_t)B * E * S * 2);   // (gamma*enc_b)^T [B,E,S]
  float*  gammaB  = (float*) alloc((size_t)B * S * 4);
  float*  gm1     = (float*) alloc((size_t)B * S * 4);
  float*  denrow  = (float*) alloc((size_t)B * T * 4);
  float*  sb      = (float*) alloc((size_t)B * 4);
  float*  midn2   = (float*) alloc((size_t)B * T * 4);

  // ---- prep ----
  prep_weight<<<E,  256, 0, stream>>>(w_in,  w_inT,  zn_in,  CC, E);
  prep_weight<<<CC, 256, 0, stream>>>(w_out, w_outT, zn_out, E,  CC);
  rows_to_bf16_sumsq<<<B * T, 256, 0, stream>>>(x, x_bf, xn2, CC);
  // enc_a [B,E,S] -> aT [B,S,E]
  transpose_scale_f32_bf16<<<dim3(S / 32, E / 32, B), 256, 0, stream>>>(
      enc_a, aT, nullptr, E, S);
  row_sumsq_bf16<<<B * S, 256, 0, stream>>>(aT, y2, E);
  gamma_kernel<<<B * S, 256, 0, stream>>>(enc_b, gammaB, gm1, E);
  // gamma*enc_b [B,S,E] -> gbT [B,E,S]
  transpose_scale_f32_bf16<<<dim3(E / 32, S / 32, B), 256, 0, stream>>>(
      enc_b, gbT, gammaB, S, E);
  mask_count<<<B, 256, 0, stream>>>(mask, sb, S);

  // ---- stage 1: x @ zu_in -> xw; q ----
  gemm_bf16_tn<<<dim3(E / 64, T / 128, B), 256, 0, stream>>>(
      x_bf, w_inT, xw, T, E, CC, (long long)T * CC, 0, (long long)T * E);
  poincare_q_kernel<<<B * T, 256, 0, stream>>>(xw, xn2, zn_in, b_in, tgt, q_bf, qn2, E);

  // ---- stage 2: q @ enc_a -> xy; masked softmax -> attn ----
  gemm_bf16_tn<<<dim3(S / 64, T / 128, B), 256, 0, stream>>>(
      q_bf, aT, xy, T, S, E, (long long)T * E, (long long)S * E, (long long)T * S);
  attn_softmax_kernel<<<B * T, 256, 0, stream>>>(
      xy, qn2, y2, mask, gm1, scale, attn_out, attn_bf, denrow, S, T);

  // ---- stage 3: attn @ (gamma*enc_b) -> nom (xw); gyromidpoint -> mid ----
  gemm_bf16_tn<<<dim3(E / 64, T / 128, B), 256, 0, stream>>>(
      attn_bf, gbT, xw, T, E, S, (long long)T * S, (long long)E * S, (long long)T * E);
  midpoint_kernel<<<B * T, 256, 0, stream>>>(xw, denrow, sb, q_bf, midn2, E, T);

  // ---- stage 4: mid @ zu_out -> mw; final mobius ops -> out ----
  gemm_bf16_tn<<<dim3(CC / 64, T / 128, B), 256, 0, stream>>>(
      q_bf, w_outT, mwf, T, CC, E, (long long)T * E, 0, (long long)T * CC);
  final_kernel<<<B * T, 256, 0, stream>>>(
      mwf, midn2, zn_out, b_out, x, xn2, out_main, CC);
}
